// Model_63556926046610
// MI455X (gfx1250) — matrix operand order repaired and checked by reading, not compiled, not run on hardware
//
#include <hip/hip_runtime.h>
#include <hip/hip_bf16.h>
#include <math.h>

typedef _Float16 h16;
typedef __attribute__((ext_vector_type(16))) _Float16 v16h;
typedef __attribute__((ext_vector_type(8)))  _Float16 v8h;
typedef __attribute__((ext_vector_type(8)))  float    v8f;

// ---------------- problem constants ----------------
#define B_      2
#define S_      2048
#define D_      512
#define H_      8
#define HD_     64
#define NBLK_   128
#define L_      16          // S_/NBLK_
#define BL_     32          // B_*L_
#define S2_     132         // NBLK_ + R(4)
#define S2P_    160         // padded to multiple of 32
#define T1_     4096        // B_*S_
#define T2_     4224        // BL_*S2_
#define TMAX_   4224
#define FFNU_   4096
#define FFNH_   2048

// ---------------- device helpers ----------------
__device__ __forceinline__ void lds_fence() {
  // per-wave DS ops are in-order; this orders the compiler and drains DScnt
  asm volatile("s_wait_dscnt 0" ::: "memory");
}

// A-fragment: 16x32 (MxK) f16, row r = lane&15; lanes<16 hold K{0..7,16..23},
// lanes>=16 hold K{8..15,24..31}  (ISA 7.12.2)
__device__ __forceinline__ v16h load_fragA(const h16* p, int ld) {
  int lane = threadIdx.x & 31;
  int r  = lane & 15;
  int kb = (lane >> 4) << 3;
  const h16* q = p + (size_t)r * ld + kb;
  union { v16h v; v8h h[2]; } u;
  u.h[0] = *(const v8h*)(q);
  u.h[1] = *(const v8h*)(q + 16);
  return u.v;
}
// B-fragment: 32x16 (KxN) f16, col n = lane&15; lanes<16 hold K{0..7,16..23},
// lanes>=16 hold K{8..15,24..31}. Source layout: row n contiguous along K (stride ld).
__device__ __forceinline__ v16h load_fragB(const h16* p, int ld) {
  int lane = threadIdx.x & 31;
  int n  = lane & 15;
  int kb = (lane >> 4) << 3;
  const h16* q = p + (size_t)n * ld + kb;
  union { v16h v; v8h h[2]; } u;
  u.h[0] = *(const v8h*)(q);
  u.h[1] = *(const v8h*)(q + 16);
  return u.v;
}
__device__ __forceinline__ v8f wmma_f16(v16h a, v16h b, v8f c) {
  return __builtin_amdgcn_wmma_f32_16x16x32_f16(false, a, false, b, (short)0, c, false, false);
}

// ---------------- kernels ----------------
__global__ void k_f2h(const float* __restrict__ in, h16* __restrict__ out, int n) {
  int i = blockIdx.x * blockDim.x + threadIdx.x;
  if (i < n) out[i] = (h16)in[i];
}

// one wave per row of 512; out = f16 normalized
__global__ __launch_bounds__(256) void k_rmsnorm(const float* __restrict__ x,
                                                 const float* __restrict__ w,
                                                 h16* __restrict__ out, int M) {
  int wv = blockIdx.x * 8 + (threadIdx.x >> 5);
  if (wv >= M) return;
  int lane = threadIdx.x & 31;
  const float* row = x + (size_t)wv * D_;
  float vals[16];
  float s = 0.f;
#pragma unroll
  for (int i = 0; i < 16; i++) { float v = row[lane + i * 32]; vals[i] = v; s += v * v; }
#pragma unroll
  for (int off = 16; off >= 1; off >>= 1) s += __shfl_xor(s, off, 32);
  float r = rsqrtf(s * (1.0f / (float)D_) + 1e-5f);
  h16* orow = out + (size_t)wv * D_;
#pragma unroll
  for (int i = 0; i < 16; i++) { int c = lane + i * 32; orow[c] = (h16)(vals[i] * r * w[c]); }
}

// C[M,N] = A[M,K] @ B[N,K]^T (+ Res), f16 in, f32 out.
// Wave computes a 32x64 tile: 2 A-frags x 4 B-frags = 8 WMMAs per 32-K step.
// K-loop steps 64 with ping-pong fragment buffers (no register rotation moves:
// each buffer is refilled while the other feeds the WMMAs). Requires K % 64 == 0.
__global__ __launch_bounds__(256) void k_gemm(const h16* __restrict__ A,
                                              const h16* __restrict__ Bw,
                                              const float* __restrict__ Res,
                                              float* __restrict__ Out,
                                              int M, int N, int K, int residual) {
  int lane = threadIdx.x & 31;
  int wv = blockIdx.x * 8 + (threadIdx.x >> 5);
  int nt64 = N >> 6;
  int tn64 = wv % nt64;
  int tm   = wv / nt64;           // 32-row tile index
  if (tm >= (M >> 5)) return;
  v8f acc[2][4];
#pragma unroll
  for (int r2 = 0; r2 < 2; r2++)
#pragma unroll
    for (int t = 0; t < 4; t++)
#pragma unroll
      for (int i = 0; i < 8; i++) acc[r2][t][i] = 0.f;

  const h16* Ap0 = A + (size_t)tm * 32 * K;
  const h16* Ap1 = Ap0 + (size_t)16 * K;
  const h16* Bp  = Bw + (size_t)(tn64 * 4) * 16 * K;

  // buffer A primed with k=0
  v16h a0A = load_fragA(Ap0, K);
  v16h a1A = load_fragA(Ap1, K);
  v16h bA[4];
#pragma unroll
  for (int t = 0; t < 4; t++) bA[t] = load_fragB(Bp + (size_t)t * 16 * K, K);
  v16h a0B, a1B, bB[4];

  for (int k0 = 0; k0 < K; k0 += 64) {
    int kb = k0 + 32;
    // refill buffer B while buffer A computes
    a0B = load_fragA(Ap0 + kb, K);
    a1B = load_fragA(Ap1 + kb, K);
#pragma unroll
    for (int t = 0; t < 4; t++) bB[t] = load_fragB(Bp + (size_t)t * 16 * K + kb, K);
#pragma unroll
    for (int t = 0; t < 4; t++) {
      acc[0][t] = wmma_f16(a0A, bA[t], acc[0][t]);
      acc[1][t] = wmma_f16(a1A, bA[t], acc[1][t]);
    }
    int kc = k0 + 64;
    if (kc < K) {                 // refill buffer A while buffer B computes
      a0A = load_fragA(Ap0 + kc, K);
      a1A = load_fragA(Ap1 + kc, K);
#pragma unroll
      for (int t = 0; t < 4; t++) bA[t] = load_fragB(Bp + (size_t)t * 16 * K + kc, K);
    }
#pragma unroll
    for (int t = 0; t < 4; t++) {
      acc[0][t] = wmma_f16(a0B, bB[t], acc[0][t]);
      acc[1][t] = wmma_f16(a1B, bB[t], acc[1][t]);
    }
  }

  int hi8 = (lane >> 4) * 8;
  int n0  = lane & 15;
#pragma unroll
  for (int r2 = 0; r2 < 2; r2++)
#pragma unroll
    for (int t = 0; t < 4; t++) {
      int n = (tn64 * 4 + t) * 16 + n0;
#pragma unroll
      for (int i = 0; i < 8; i++) {
        size_t off = (size_t)(tm * 32 + r2 * 16 + hi8 + i) * N + n;
        float v = acc[r2][t][i];
        if (residual) v += Res[off];
        Out[off] = v;
      }
    }
}

// qkv f32 [NS*SL,1536] -> rope'd q,k f16 [NS,H,SP,64]; v transposed [NS,H,64,SP]
__global__ void k_rope(const float* __restrict__ qkv, h16* __restrict__ qf,
                       h16* __restrict__ kf, h16* __restrict__ vT,
                       int NS, int SL, int SP) {
  int idx = blockIdx.x * blockDim.x + threadIdx.x;
  int i = idx & 31;
  int t = idx >> 5;
  int s = t % SP; t /= SP;
  int h = t & 7;  int seq = t >> 3;
  if (seq >= NS) return;
  float q1 = 0.f, q2 = 0.f, k1 = 0.f, k2 = 0.f, v1 = 0.f, v2 = 0.f;
  if (s < SL) {
    const float* rp = qkv + ((size_t)seq * SL + s) * 1536 + h * 64;
    float a1 = rp[i],        a2 = rp[32 + i];
    float b1 = rp[512 + i],  b2 = rp[512 + 32 + i];
    v1 = rp[1024 + i];       v2 = rp[1024 + 32 + i];
    float inv = powf(10000.0f, -(float)i / 32.0f);
    float fr  = (float)s * inv;
    float c = cosf(fr), sn = sinf(fr);
    q1 =  a1 * c + a2 * sn;  q2 = -a1 * sn + a2 * c;
    k1 =  b1 * c + b2 * sn;  k2 = -b1 * sn + b2 * c;
  }
  size_t ob = ((size_t)(seq * H_ + h) * SP + s) * 64;
  size_t vb = (size_t)(seq * H_ + h) * 64;
  qf[ob + i] = (h16)q1;  qf[ob + 32 + i] = (h16)q2;
  kf[ob + i] = (h16)k1;  kf[ob + 32 + i] = (h16)k2;
  vT[(vb + i)      * SP + s] = (h16)v1;
  vT[(vb + 32 + i) * SP + s] = (h16)v2;
}

// flash attention, one wave per (seq, head, 16-query tile)
__global__ __launch_bounds__(128) void k_attn(const h16* __restrict__ qf,
                                              const h16* __restrict__ kf,
                                              const h16* __restrict__ vT,
                                              const int* __restrict__ docp,
                                              h16* __restrict__ ao,
                                              int NS, int SL, int SP, int QT,
                                              int blockCausal) {
  __shared__ float sc[4][16][32];
  __shared__ __align__(16) h16 pb[4][16][32];
  __shared__ float al[4][16];
  __shared__ float lr[4][16];
  int w    = threadIdx.x >> 5;
  int lane = threadIdx.x & 31;
  int gid  = blockIdx.x * 4 + w;
  if (gid >= NS * H_ * QT) return;
  int qt  = gid % QT;
  int h   = (gid / QT) & 7;
  int seq = gid / (QT * H_);
  int qbase = qt * 16;
  const h16* qp = qf + ((size_t)(seq * H_ + h) * SP) * 64;
  const h16* kp = kf + ((size_t)(seq * H_ + h) * SP) * 64;
  const h16* vp = vT + ((size_t)(seq * H_ + h) * 64) * SP;
  v16h qa0 = load_fragA(qp + (size_t)qbase * 64, 64);
  v16h qa1 = load_fragA(qp + (size_t)qbase * 64 + 32, 64);
  v8f acc[4];
#pragma unroll
  for (int t = 0; t < 4; t++)
#pragma unroll
    for (int i = 0; i < 8; i++) acc[t][i] = 0.f;
  float mo = -INFINITY, lo = 0.f;
  int docq = 0;
  if (docp && lane < 16) docq = docp[seq * S_ + qbase + lane];
  int kmax, klimit;
  if (blockCausal) { kmax = ((qbase / NBLK_) + 1) * NBLK_; klimit = kmax; }
  else             { klimit = SL; kmax = ((SL + 31) / 32) * 32; }
  int hi8 = (lane >> 4) * 8;

  for (int k0 = 0; k0 < kmax; k0 += 32) {
    // ---- scores: two 16x16 tiles (keys k0..k0+31), hd=64 -> 2 K-steps ----
#pragma unroll
    for (int tile = 0; tile < 2; tile++) {
      v8f s0;
#pragma unroll
      for (int i = 0; i < 8; i++) s0[i] = 0.f;
      const h16* kt = kp + (size_t)(k0 + tile * 16) * 64;
      s0 = wmma_f16(qa0, load_fragB(kt, 64), s0);
      s0 = wmma_f16(qa1, load_fragB(kt + 32, 64), s0);
#pragma unroll
      for (int i = 0; i < 8; i++) sc[w][hi8 + i][tile * 16 + (lane & 15)] = s0[i];
    }
    lds_fence();
    // ---- per-row online softmax (lanes 0..15, one row each) ----
    if (lane < 16) {
      int r = lane;
      float mn = mo;
      for (int j = 0; j < 32; j++) {
        int kk = k0 + j;
        bool valid = (kk < klimit) && (!docp || docq == docp[seq * S_ + kk]);
        if (valid) mn = fmaxf(mn, sc[w][r][j] * 0.125f);
      }
      float alpha = (mo == -INFINITY) ? 0.f : expf(mo - mn);
      float ls = alpha * lo;
      for (int j = 0; j < 32; j++) {
        int kk = k0 + j;
        bool valid = (kk < klimit) && (!docp || docq == docp[seq * S_ + kk]);
        float p = valid ? expf(sc[w][r][j] * 0.125f - mn) : 0.f;
        pb[w][r][j] = (h16)p;
        ls += p;
      }
      mo = mn; lo = ls; al[w][r] = alpha;
    }
    lds_fence();
    // ---- rescale accumulators, then O += P @ V (K=32) ----
#pragma unroll
    for (int t = 0; t < 4; t++)
#pragma unroll
      for (int i = 0; i < 8; i++) acc[t][i] *= al[w][hi8 + i];
    v16h pf = load_fragA(&pb[w][0][0], 32);
#pragma unroll
    for (int t = 0; t < 4; t++) {
      v16h vb = load_fragB(vp + (size_t)(t * 16) * SP + k0, SP);
      acc[t] = wmma_f16(pf, vb, acc[t]);
    }
    lds_fence();
  }
  if (lane < 16) lr[w][lane] = lo;
  lds_fence();
  int n0 = lane & 15;
#pragma unroll
  for (int t = 0; t < 4; t++)
#pragma unroll
    for (int i = 0; i < 8; i++) {
      int m = hi8 + i;
      int qg = qbase + m;
      if (qg < SL) {
        float linv = 1.0f / lr[w][m];
        ao[((size_t)seq * SL + qg) * D_ + h * 64 + t * 16 + n0] = (h16)(acc[t][i] * linv);
      }
    }
}

__global__ void k_gate(const float* __restrict__ up, h16* __restrict__ g, int M) {
  int idx = blockIdx.x * blockDim.x + threadIdx.x;
  if (idx >= M * FFNH_) return;
  int m = idx / FFNH_, j = idx % FFNH_;
  float u = up[(size_t)m * FFNU_ + j];
  float v = up[(size_t)m * FFNU_ + FFNH_ + j];
  float si = u / (1.0f + expf(-u));
  g[idx] = (h16)(si * v);
}

__global__ void k_xrinit(const float* __restrict__ hx, const float* __restrict__ rtk,
                         float* __restrict__ x2) {
  int idx = blockIdx.x * blockDim.x + threadIdx.x;
  if (idx >= T2_ * D_) return;
  int c = idx % D_;
  int row = idx / D_;
  int s = row % S2_, bl = row / S2_;
  int b = bl >> 4, l = bl & 15;
  float v = (s < NBLK_) ? hx[((size_t)(b * S_ + l * NBLK_ + s)) * D_ + c]
                        : rtk[(s - NBLK_) * D_ + c];
  x2[idx] = v;
}

__global__ void k_gather(const float* __restrict__ x2, h16* __restrict__ rp) {
  int idx = blockIdx.x * blockDim.x + threadIdx.x;
  if (idx >= 128 * D_) return;
  int c = idx % D_, rr = idx / D_;
  int bl = rr >> 2, r = rr & 3;
  rp[idx] = (h16)x2[((size_t)(bl * S2_ + NBLK_ + r)) * D_ + c];
}

// logits[g,t,e] = sum_d xro[g,t,d]*keys[g,d,e] with repeat/roll index mapping
__global__ void k_router(const float* __restrict__ outr, const float* __restrict__ keys,
                         float* __restrict__ logits) {
  int e = threadIdx.x;           // 0..63
  int gt = blockIdx.x;           // 0..1023
  int g = gt >> 5, t = gt & 31;
  int b = t >> 4, l = t & 15;
  int lsrc = (l + 15) & 15;      // roll(+1) along L
  int r = g >> 3;                // repeat(RM=8)
  const float* xrow = outr + (size_t)((b * 16 + lsrc) * 4 + r) * 128;  // first 64 cols used
  const float* kp = keys + (size_t)g * 64 * 64;
  float s = 0.f;
  for (int d = 0; d < 64; d++) s += xrow[d] * kp[d * 64 + e];
  logits[((size_t)g * 32 + t) * 64 + e] = s;
}

__global__ void k_topk(const float* __restrict__ logits, float* __restrict__ out) {
  int g = blockIdx.x, t = threadIdx.x;
  const float* p = logits + ((size_t)g * 32 + t) * 64;
  float m1 = -INFINITY; int i1 = 0;
  for (int e = 0; e < 64; e++) { float v = p[e]; if (v > m1) { m1 = v; i1 = e; } }
  float m2 = -INFINITY; int i2 = 0;
  for (int e = 0; e < 64; e++) { if (e == i1) continue; float v = p[e]; if (v > m2) { m2 = v; i2 = e; } }
  out[((size_t)g * 32 + t) * 2 + 0] = m1;
  out[((size_t)g * 32 + t) * 2 + 1] = m2;
  float* oi = out + 2048;
  int ln = g >> 2, gn = g & 3;
  oi[ln * 256 + t * 8 + gn * 2 + 0] = (float)i1;
  oi[ln * 256 + t * 8 + gn * 2 + 1] = (float)i2;
}

// ---------------- host orchestration ----------------
extern "C" void kernel_launch(void* const* d_in, const int* in_sizes, int n_in,
                              void* d_out, int out_size, void* d_ws, size_t ws_size,
                              hipStream_t stream) {
  const float* x      = (const float*)d_in[0];
  const int*   doc    = (const int*)  d_in[1];
  const float* encA   = (const float*)d_in[2];
  const float* encO   = (const float*)d_in[3];
  const float* encU   = (const float*)d_in[4];
  const float* encDw  = (const float*)d_in[5];
  const float* encN1  = (const float*)d_in[6];
  const float* encN2  = (const float*)d_in[7];
  const float* rtA    = (const float*)d_in[8];
  const float* rtO    = (const float*)d_in[9];
  const float* rtU    = (const float*)d_in[10];
  const float* rtDw   = (const float*)d_in[11];
  const float* rtN1   = (const float*)d_in[12];
  const float* rtN2   = (const float*)d_in[13];
  const float* rtok   = (const float*)d_in[14];
  const float* outw   = (const float*)d_in[15];
  const float* keys   = (const float*)d_in[16];
  float* ov = (float*)d_out;

  size_t cur = 0;
  auto carve = [&](size_t bytes) -> char* {
    char* p = (char*)d_ws + cur;
    cur += (bytes + 255) & ~(size_t)255;
    return p;
  };
  const size_t AW = (size_t)2 * 1536 * 512, OW = (size_t)2 * 512 * 512;
  const size_t UW = (size_t)2 * 4096 * 512, DW = (size_t)2 * 512 * 2048;
  h16* wEncA = (h16*)carve(AW * 2); h16* wEncO = (h16*)carve(OW * 2);
  h16* wEncU = (h16*)carve(UW * 2); h16* wEncD = (h16*)carve(DW * 2);
  h16* wRtA  = (h16*)carve(AW * 2); h16* wRtO  = (h16*)carve(OW * 2);
  h16* wRtU  = (h16*)carve(UW * 2); h16* wRtD  = (h16*)carve(DW * 2);
  h16* wOut  = (h16*)carve((size_t)128 * 512 * 2);
  float* hx  = (float*)carve((size_t)T1_ * D_ * 4);
  float* x2  = (float*)carve((size_t)T2_ * D_ * 4);
  h16*  hn   = (h16*) carve((size_t)TMAX_ * D_ * 2);
  h16*  ao   = (h16*) carve((size_t)TMAX_ * D_ * 2);
  h16*  gbuf = (h16*) carve((size_t)TMAX_ * FFNH_ * 2);
  float* big = (float*)carve((size_t)TMAX_ * FFNU_ * 4);   // qkv / up union
  const size_t QKVPAD = (size_t)BL_ * H_ * S2P_ * HD_;     // >= enc size
  h16* qf = (h16*)carve(QKVPAD * 2);
  h16* kf = (h16*)carve(QKVPAD * 2);
  h16* vT = (h16*)carve(QKVPAD * 2);
  h16*  routp  = (h16*) carve((size_t)128 * 512 * 2);
  float* outr  = (float*)carve((size_t)128 * 128 * 4);
  float* logits= (float*)carve((size_t)32 * 32 * 64 * 4);
  (void)ws_size; (void)n_in; (void)in_sizes; (void)out_size;

  auto f2h = [&](const float* src, h16* dst, size_t n) {
    k_f2h<<<(int)((n + 255) / 256), 256, 0, stream>>>(src, dst, (int)n);
  };
  f2h(encA, wEncA, AW); f2h(encO, wEncO, OW); f2h(encU, wEncU, UW); f2h(encDw, wEncD, DW);
  f2h(rtA,  wRtA,  AW); f2h(rtO,  wRtO,  OW); f2h(rtU,  wRtU,  UW); f2h(rtDw,  wRtD,  DW);
  f2h(outw, wOut, (size_t)128 * 512);

  hipMemcpyAsync(hx, x, (size_t)T1_ * D_ * 4, hipMemcpyDeviceToDevice, stream);

  auto gemm = [&](const h16* A, const h16* Bw, const float* Res, float* Out,
                  int M, int N, int K, int residual) {
    int waves = (M / 32) * (N / 64);
    k_gemm<<<(waves + 7) / 8, 256, 0, stream>>>(A, Bw, Res, Out, M, N, K, residual);
  };

  auto layer = [&](float* xb, int M, int NS, int SL, int SP,
                   const h16* aw, const h16* aow, const h16* uw, const h16* dw,
                   const float* n1, const float* n2, const int* docp, int blockCausal) {
    int QT = (SL + 15) / 16;
    float* qkv = big;
    float* up  = big;
    // 1. rmsnorm -> hn
    k_rmsnorm<<<(M + 7) / 8, 256, 0, stream>>>(xb, n1, hn, M);
    // 2. qkv
    gemm(hn, aw, nullptr, qkv, M, 1536, 512, 0);
    // 3. rope + layout
    {
      int nt = NS * H_ * SP * 32;
      k_rope<<<(nt + 255) / 256, 256, 0, stream>>>(qkv, qf, kf, vT, NS, SL, SP);
    }
    // 4. attention -> ao (f16, token-major)
    {
      int tw = NS * H_ * QT;
      k_attn<<<(tw + 3) / 4, 128, 0, stream>>>(qf, kf, vT, docp, ao, NS, SL, SP, QT, blockCausal);
    }
    // 5. x += ao @ aow^T
    gemm(ao, aow, xb, xb, M, 512, 512, 1);
    // 6. rmsnorm -> hn
    k_rmsnorm<<<(M + 7) / 8, 256, 0, stream>>>(xb, n2, hn, M);
    // 7. up
    gemm(hn, uw, nullptr, up, M, FFNU_, 512, 0);
    // 8. gate
    {
      int nt = M * FFNH_;
      k_gate<<<(nt + 255) / 256, 256, 0, stream>>>(up, gbuf, M);
    }
    // 9. x += g @ dw^T
    gemm(gbuf, dw, xb, xb, M, 512, FFNH_, 1);
  };

  // encoder layers
  for (int i = 0; i < 2; i++) {
    layer(hx, T1_, B_, S_, S_,
          wEncA + (size_t)i * 1536 * 512, wEncO + (size_t)i * 512 * 512,
          wEncU + (size_t)i * 4096 * 512, wEncD + (size_t)i * 512 * 2048,
          encN1 + i * 512, encN2 + i * 512, doc, 1);
  }
  // build router sequences
  {
    int nt = T2_ * D_;
    k_xrinit<<<(nt + 255) / 256, 256, 0, stream>>>(hx, rtok, x2);
  }
  // router-token layers
  for (int i = 0; i < 2; i++) {
    layer(x2, T2_, BL_, S2_, S2P_,
          wRtA + (size_t)i * 1536 * 512, wRtO + (size_t)i * 512 * 512,
          wRtU + (size_t)i * 4096 * 512, wRtD + (size_t)i * 512 * 2048,
          rtN1 + i * 512, rtN2 + i * 512, nullptr, 0);
  }
  // head: gather router tokens, project, route, top-2
  {
    int nt = 128 * D_;
    k_gather<<<(nt + 255) / 256, 256, 0, stream>>>(x2, routp);
  }
  gemm(routp, wOut, nullptr, outr, 128, 128, 512, 0);
  k_router<<<32 * 32, 64, 0, stream>>>(outr, keys, logits);
  k_topk<<<32, 32, 0, stream>>>(logits, ov);
}
